// RNN_Build_11879879542664
// MI455X (gfx1250) — matrix operand order repaired and checked by reading, not compiled, not run on hardware
//
#include <hip/hip_runtime.h>
#include <math.h>

// Problem constants (from reference)
#define BB 4096
#define TT 80
#define VV 10000
#define EE 100
#define HH 64
#define EPAD 128

#define AS1 __attribute__((address_space(1)))
#define AS3 __attribute__((address_space(3)))

typedef __attribute__((ext_vector_type(16))) _Float16 v16h;
typedef __attribute__((ext_vector_type(8)))  _Float16 v8h;
typedef __attribute__((ext_vector_type(4)))  _Float16 v4h;
typedef __attribute__((ext_vector_type(2)))  _Float16 v2h;
typedef __attribute__((ext_vector_type(8)))  float    v8f;
typedef int v2i_g __attribute__((vector_size(8)));   // matches builtin's b64 payload type

#if defined(__gfx1250__) && __has_builtin(__builtin_amdgcn_global_load_async_to_lds_b64)
#define HAVE_ASYNC_LDS 1
#else
#define HAVE_ASYNC_LDS 0
#endif

__device__ __forceinline__ v8f wmma_f16(v16h a, v16h b, v8f c) {
    return __builtin_amdgcn_wmma_f32_16x16x32_f16(
        /*neg_a=*/false, a, /*neg_b=*/false, b,
        /*c_mod=*/(short)0, c, /*reuse_a=*/false, /*reuse_b=*/false);
}

// Hardware tanh: single TRANS32 op on gfx1250 (co-executes with WMMA).
__device__ __forceinline__ float fast_tanh(float x) {
#if defined(__gfx1250__)
#if __has_builtin(__builtin_amdgcn_tanhf)
    return __builtin_amdgcn_tanhf(x);
#else
    float y;
    asm("v_tanh_f32 %0, %1" : "=v"(y) : "v"(x));
    return y;
#endif
#else
    return tanhf(x);
#endif
}

// A-fragment (16x32 f16) from row-major LDS [16][ld]:
// lane L: M = L&15 ; lanes 0-15: K = kb..kb+7 (VGPR0-3), kb+16..kb+23 (VGPR4-7);
// lanes 16-31: kb += 8.  kb base = kc*32.
__device__ __forceinline__ v16h load_a_frag(const _Float16* src, int ld, int lane, int kc) {
    int m  = lane & 15;
    int kb = kc * 32 + ((lane >> 4) << 3);
    v8h lo = *(const v8h*)(src + m * ld + kb);
    v8h hi = *(const v8h*)(src + m * ld + kb + 16);
    v16h r;
#pragma unroll
    for (int i = 0; i < 8; ++i) { r[i] = lo[i]; r[8 + i] = hi[i]; }
    return r;
}

// B-fragment (32x16 f16) from transposed weights wT stored [N][ldk] (K contiguous):
// lane L: N = L&15 ; lanes 0-15: K = kb..kb+7, kb+16..kb+23 ; lanes 16-31: kb += 8 (kb = kc*32)
__device__ __forceinline__ v16h load_b_frag(const _Float16* wT, int ldk, int lane, int kc) {
    int n  = lane & 15;
    int kb = kc * 32 + ((lane >> 4) << 3);
    v8h lo = *(const v8h*)(wT + n * ldk + kb);
    v8h hi = *(const v8h*)(wT + n * ldk + kb + 16);
    v16h r;
#pragma unroll
    for (int i = 0; i < 8; ++i) { r[i] = lo[i]; r[8 + i] = hi[i]; }
    return r;
}

// One RNN layer for a 16-row tile, weights in registers.
//   hbuf = tanh(xsrc @ Wx + hbuf @ Wh [+ bias])
// Accumulators start from literal-zero C (inline SRC2=0 in the first WMMA).
template <int NKX, bool ADD_BIAS>
__device__ __forceinline__ void rnn_layer(const _Float16* xsrc, int ldx,
                                          _Float16* hbuf,
                                          const v16h wx[][4],  // [NKX][4]
                                          const v16h wh[][4],  // [2][4]
                                          const float* bias, int lane) {
    v8f acc[4];
    {
        v8f zc = {0.f, 0.f, 0.f, 0.f, 0.f, 0.f, 0.f, 0.f};
        v16h a0 = load_a_frag(xsrc, ldx, lane, 0);
#pragma unroll
        for (int n = 0; n < 4; ++n) acc[n] = wmma_f16(a0, wx[0][n], zc);
    }
#pragma unroll
    for (int kc = 1; kc < NKX; ++kc) {
        v16h a = load_a_frag(xsrc, ldx, lane, kc);
#pragma unroll
        for (int n = 0; n < 4; ++n) acc[n] = wmma_f16(a, wx[kc][n], acc[n]);
    }
#pragma unroll
    for (int kc = 0; kc < 2; ++kc) {
        v16h a = load_a_frag(hbuf, HH, lane, kc);
#pragma unroll
        for (int n = 0; n < 4; ++n) acc[n] = wmma_f16(a, wh[kc][n], acc[n]);
    }
    asm volatile("" ::: "memory");  // all hbuf reads issued before stores (LDS in-order)
    int m0 = (lane >> 4) << 3;      // 0 or 8
#pragma unroll
    for (int n = 0; n < 4; ++n) {
#pragma unroll
        for (int r = 0; r < 8; ++r) {
            float v = acc[n][r];
            if constexpr (ADD_BIAS) v += bias[n];
            hbuf[(m0 + r) * HH + n * 16 + (lane & 15)] = (_Float16)fast_tanh(v);
        }
    }
    asm volatile("s_wait_dscnt 0" ::: "memory");
}

// Stage one embedding row tile (16 rows x E) into an LDS x-buffer; 2 lanes per row.
template <bool EMB16>
__device__ __forceinline__ void stage_x(const void* embv, int token, _Float16* dstRow, int q) {
    if constexpr (EMB16) {
        const _Float16* srow = (const _Float16*)embv + (size_t)token * EE;
#if HAVE_ASYNC_LDS
#pragma unroll
        for (int j = q; j < 25; j += 2)   // 25 x b64 chunks per row, split odd/even
            __builtin_amdgcn_global_load_async_to_lds_b64(
                (AS1 v2i_g*)(srow + 4 * j), (AS3 v2i_g*)(dstRow + 4 * j), 0, 0);
#else
#pragma unroll
        for (int j = q; j < 25; j += 2) *(v4h*)(dstRow + 4 * j) = *(const v4h*)(srow + 4 * j);
#endif
    } else {
        const float* srow = (const float*)embv + (size_t)token * EE;
        const float2* src = (const float2*)(srow + q * 50);
        _Float16* d2 = dstRow + q * 50;
#pragma unroll
        for (int j = 0; j < 25; ++j) {
            float2 v = src[j];
            v2h p; p[0] = (_Float16)v.x; p[1] = (_Float16)v.y;
            *(v2h*)(d2 + 2 * j) = p;
        }
    }
}

__device__ __forceinline__ void wait_stage() {
#if HAVE_ASYNC_LDS
    asm volatile("s_wait_asynccnt 0" ::: "memory");
#endif
    asm volatile("s_wait_dscnt 0" ::: "memory");
}

// One-time emb table conversion f32 -> f16 (into d_ws).
__global__ __launch_bounds__(256) void emb_to_f16_kernel(const float* __restrict__ emb,
                                                         _Float16* __restrict__ out, int n4) {
    int i = blockIdx.x * blockDim.x + threadIdx.x;
    if (i < n4) {
        float4 v = ((const float4*)emb)[i];
        v4h o;
        o[0] = (_Float16)v.x; o[1] = (_Float16)v.y;
        o[2] = (_Float16)v.z; o[3] = (_Float16)v.w;
        ((v4h*)out)[i] = o;
    }
}

template <bool EMB16>
__global__ __launch_bounds__(64) void rnn2_wmma_kernel(
    const int* __restrict__ tokens, const void* __restrict__ embv,
    const float* __restrict__ Wx0, const float* __restrict__ Wh0, const float* __restrict__ b0,
    const float* __restrict__ Wx1, const float* __restrict__ Wh1, const float* __restrict__ b1,
    const float* __restrict__ Wout, const float* __restrict__ bout,
    float* __restrict__ out) {

    // Single LDS arena, two phases:
    //  phase 1: transposed f16 weights  (Wx0T 8192 | Wh0T 4096 | Wx1T 4096 | Wh1T 4096 halves)
    //  phase 2: per-wave double-buffered x-stage [2][16][128] + h0/h1 [16][64] (overlaid)
    __shared__ __align__(16) _Float16 smem[20480];  // 40 KB

    const int tid  = threadIdx.x;
    const int lane = tid & 31;
    const int w    = tid >> 5;
    const int rowBase = blockIdx.x * 32 + w * 16;

    // ---- phase 1: stage transposed weights into LDS; b0 folded into column k==100 ----
    _Float16* sWx0 = smem;           // [64][128]: k<100 = Wx0^T, k==100 = b0, rest 0
    _Float16* sWh0 = smem + 8192;    // [64][64]
    _Float16* sWx1 = smem + 12288;   // [64][64]
    _Float16* sWh1 = smem + 16384;   // [64][64]
    for (int i = tid; i < HH * EPAD; i += 64) {
        int n = i >> 7, k = i & (EPAD - 1);
        float v = (k < EE) ? Wx0[k * HH + n] : ((k == EE) ? b0[n] : 0.0f);
        sWx0[i] = (_Float16)v;
    }
    for (int i = tid; i < HH * HH; i += 64) {
        int n = i >> 6, k = i & 63;
        sWh0[i] = (_Float16)Wh0[k * HH + n];
        sWx1[i] = (_Float16)Wx1[k * HH + n];
        sWh1[i] = (_Float16)Wh1[k * HH + n];
    }
    __syncthreads();

    // ---- hoist all weight fragments into persistent VGPRs (loop-invariant) ----
    v16h wx0f[4][4], wh0f[2][4], wx1f[2][4], wh1f[2][4];
#pragma unroll
    for (int kc = 0; kc < 4; ++kc)
#pragma unroll
        for (int n = 0; n < 4; ++n)
            wx0f[kc][n] = load_b_frag(sWx0 + (n * 16) * EPAD, EPAD, lane, kc);
#pragma unroll
    for (int kc = 0; kc < 2; ++kc)
#pragma unroll
        for (int n = 0; n < 4; ++n) {
            wh0f[kc][n] = load_b_frag(sWh0 + (n * 16) * HH, HH, lane, kc);
            wx1f[kc][n] = load_b_frag(sWx1 + (n * 16) * HH, HH, lane, kc);
            wh1f[kc][n] = load_b_frag(sWh1 + (n * 16) * HH, HH, lane, kc);
        }
    float bias1[4];
#pragma unroll
    for (int n = 0; n < 4; ++n) bias1[n] = b1[n * 16 + (lane & 15)];
    asm volatile("s_wait_dscnt 0" ::: "memory");
    __syncthreads();  // all waves done reading weights before overlay reuse

    // ---- phase 2: overlay working buffers on the same LDS ----
    _Float16* xb0 = smem + w * 2048;          // [16][128] buffer 0
    _Float16* xb1 = smem + 4096 + w * 2048;   // [16][128] buffer 1
    _Float16* h0w = smem + 8192 + w * 1024;   // [16][64]
    _Float16* h1w = smem + 10240 + w * 1024;  // [16][64]
    {
        v8h z;
#pragma unroll
        for (int i = 0; i < 8; ++i) z[i] = (_Float16)0.0f;
        for (int i = lane; i < 256; i += 32) *(v8h*)(xb0 + i * 8) = z;
        for (int i = lane; i < 256; i += 32) *(v8h*)(xb1 + i * 8) = z;
        for (int i = lane; i < 128; i += 32) *(v8h*)(h0w + i * 8) = z;
        for (int i = lane; i < 128; i += 32) *(v8h*)(h1w + i * 8) = z;
    }
    if (lane < 16) {  // constant-1 column -> adds b0 through the GEMM
        xb0[lane * EPAD + EE] = (_Float16)1.0f;
        xb1[lane * EPAD + EE] = (_Float16)1.0f;
    }
    asm volatile("s_wait_dscnt 0" ::: "memory");

    // ---- software-pipelined recurrence: gather x(t+1) async while computing x(t) ----
    const int rowS = lane >> 1;   // 2 lanes stage one row
    const int q    = lane & 1;
    const int* trow = tokens + (size_t)(rowBase + rowS) * TT;

    _Float16* cur = xb0;
    _Float16* nxt = xb1;
    stage_x<EMB16>(embv, trow[0], cur + rowS * EPAD, q);  // prologue: x(0)
    wait_stage();

    for (int t = 0; t < TT; ++t) {
        if (t + 1 < TT)  // issue gather for next step; completes behind the 40 WMMAs below
            stage_x<EMB16>(embv, trow[t + 1], nxt + rowS * EPAD, q);

        rnn_layer<4, false>(cur, EPAD, h0w, wx0f, wh0f, nullptr, lane);  // bias via k=100 col
        rnn_layer<2, true >(h0w, HH,  h1w, wx1f, wh1f, bias1,  lane);

        wait_stage();  // x(t+1) resident before next iteration
        _Float16* tmp = cur; cur = nxt; nxt = tmp;
    }

    // ---- output head: sigmoid(h1 @ Wout + bout), one row per lane 0..15 ----
    asm volatile("s_wait_dscnt 0" ::: "memory");
    if (lane < 16) {
        float acc = 0.0f;
#pragma unroll
        for (int k = 0; k < HH; ++k)
            acc += (float)h1w[lane * HH + k] * Wout[k];
        float z = acc + bout[0];
        out[rowBase + lane] = 1.0f / (1.0f + expf(-z));
    }
}

extern "C" void kernel_launch(void* const* d_in, const int* in_sizes, int n_in,
                              void* d_out, int out_size, void* d_ws, size_t ws_size,
                              hipStream_t stream) {
    const int*   tokens = (const int*)d_in[0];
    const float* emb    = (const float*)d_in[1];
    const float* Wx0    = (const float*)d_in[2];
    const float* Wh0    = (const float*)d_in[3];
    const float* b0     = (const float*)d_in[4];
    const float* Wx1    = (const float*)d_in[5];
    const float* Wh1    = (const float*)d_in[6];
    const float* b1     = (const float*)d_in[7];
    const float* Wout   = (const float*)d_in[8];
    const float* bout   = (const float*)d_in[9];
    float* out = (float*)d_out;

    dim3 grid(BB / 32);   // 128 blocks, each = 2 waves x 16 batch rows
    dim3 block(64);

    const size_t embBytes = (size_t)VV * EE * sizeof(_Float16);  // 2 MB
    if (ws_size >= embBytes) {
        // Pre-convert embedding table to f16: halves gather traffic, enables async-LDS staging.
        _Float16* emb16 = (_Float16*)d_ws;
        int n4 = VV * EE / 4;
        emb_to_f16_kernel<<<(n4 + 255) / 256, 256, 0, stream>>>(emb, emb16, n4);
        rnn2_wmma_kernel<true><<<grid, block, 0, stream>>>(tokens, (const void*)emb16,
                                                           Wx0, Wh0, b0, Wx1, Wh1, b1,
                                                           Wout, bout, out);
    } else {
        rnn2_wmma_kernel<false><<<grid, block, 0, stream>>>(tokens, (const void*)emb,
                                                            Wx0, Wh0, b0, Wx1, Wh1, b1,
                                                            Wout, bout, out);
    }
}
